// WindowAttention1D_83382495084931
// MI455X (gfx1250) — matrix operand order repaired and checked by reading, not compiled, not run on hardware
//
#include <hip/hip_runtime.h>

// ---------------------------------------------------------------------------
// CDNA5 (gfx1250) WindowAttention1D: bf16 WMMA pipeline with fp32 accumulate.
//   K1: qkv = x @ w_qkv + b_qkv      (fp32 in -> bf16 out, WMMA bf16, dbl-buf)
//   K2: per-(window,head) attention  (WMMA bf16, register softmax,
//                                     async global->LDS staging of K)
//   K3: out = attn @ w_proj + b_proj (bf16 in -> fp32 out, WMMA bf16,
//                                     async global->LDS staging of A, dbl-buf)
// ---------------------------------------------------------------------------

typedef __attribute__((ext_vector_type(16))) __bf16 v16bf;
typedef __attribute__((ext_vector_type(8)))  float  v8f;

constexpr int LDA = 40;   // 80B pitch: lane->bank map is a 16-perm (conflict-free)
constexpr int LDB = 40;

union FragBF {
  v16bf v;
  uint4 q[2];
};

__device__ __forceinline__ v8f zero8() {
  v8f z;
#pragma unroll
  for (int i = 0; i < 8; ++i) z[i] = 0.0f;
  return z;
}

__device__ __forceinline__ unsigned short f2bf(float x) {
  union { float f; unsigned u; } a; a.f = x;
  unsigned r = a.u + 0x7fffu + ((a.u >> 16) & 1u);   // round-to-nearest-even
  return (unsigned short)(r >> 16);
}

// Async 16B global -> LDS copy (GLOBAL_LOAD_ASYNC_TO_LDS_B128, ASYNCcnt).
// lds_off: wave-relative LDS byte offset (low 32 bits of a generic LDS ptr).
__device__ __forceinline__ void async_cp16(unsigned lds_off, const void* gsrc) {
  asm volatile("global_load_async_to_lds_b128 %0, %1, off"
               :: "v"(lds_off), "v"(gsrc) : "memory");
}

__device__ __forceinline__ void wait_async0() {
#if __has_builtin(__builtin_amdgcn_s_wait_asynccnt)
  __builtin_amdgcn_s_wait_asynccnt(0);
#else
  asm volatile("s_wait_asynccnt 0" ::: "memory");
#endif
}

__device__ __forceinline__ unsigned lds_off_u32(const void* p) {
  return (unsigned)(uintptr_t)p;   // generic LDS ptr: low 32 bits = LDS offset
}

// A fragment: 16x32 bf16 tile (MxK), row-major, row pitch `ld` halfwords.
// ISA layout: lane m = l&15 holds row m; lanes 0-15 K{0..7,16..23},
// lanes 16-31 K{8..15,24..31}  ->  kb = (l>>4)*8.
__device__ __forceinline__ v16bf load_fragA(const unsigned short* base, int ld) {
  const int lane = threadIdx.x & 31;
  const int m = lane & 15;
  const int kb = (lane >> 4) << 3;
  const unsigned short* row = base + m * ld;
  FragBF f;
  f.q[0] = *(const uint4*)(row + kb);
  f.q[1] = *(const uint4*)(row + 16 + kb);
  return f.v;
}

// B fragment: 32x16 (KxN) tile supplied N-major (i.e. B^T rows), pitch `ld`.
// Lane n = l&15; lanes 0-15 hold K{0..7,16..23},
// lanes 16-31 hold K{8..15,24..31} (same k order as A).
__device__ __forceinline__ v16bf load_fragB(const unsigned short* base, int ld) {
  const int lane = threadIdx.x & 31;
  const int n = lane & 15;
  const int kb = (lane >> 4) << 3;
  const unsigned short* row = base + n * ld + kb;
  FragBF f;
  f.q[0] = *(const uint4*)(row);
  f.q[1] = *(const uint4*)(row + 16);
  return f.v;
}

__device__ __forceinline__ v8f wmma_bf16(v16bf a, v16bf b, v8f c) {
  return __builtin_amdgcn_wmma_f32_16x16x32_bf16(
      false, a, false, b, (short)0, c, false, false);
}

// ---------------------------------------------------------------------------
// Stage one K-slice (A: 128x32 rows m0.., B: 32x128 cols n0.., transposed).
// ---------------------------------------------------------------------------
template<int N, bool ABF16>
__device__ __forceinline__ void stage_tiles(
    const float* __restrict__ Af, const unsigned short* __restrict__ Ab,
    const float* __restrict__ Wt,
    unsigned short* sA, unsigned short* sB,
    int m0, int n0, int k0, int tid) {
  constexpr int K = 512;
  if constexpr (ABF16) {
    // pure bf16 copy -> async DMA, 512 x 16B granules, 2 per thread
#pragma unroll
    for (int i = 0; i < 2; ++i) {
      int idx = i * 256 + tid;
      int r = idx >> 2, c8 = (idx & 3) << 3;
      async_cp16(lds_off_u32(sA + r * LDA + c8),
                 Ab + (size_t)(m0 + r) * K + (k0 + c8));
    }
  } else {
    // fp32 -> bf16 convert staging (VALU)
#pragma unroll
    for (int i = 0; i < 16; ++i) {
      int idx = i * 256 + tid;
      int r = idx >> 5, c = idx & 31;
      sA[r * LDA + c] = f2bf(Af[(size_t)(m0 + r) * K + (k0 + c)]);
    }
  }
  // B tile 32x128, fp32 -> bf16, transposed into sB[n][k]
#pragma unroll
  for (int i = 0; i < 16; ++i) {
    int idx = i * 256 + tid;
    int c = idx & 127, r = idx >> 7;
    sB[c * LDB + r] = f2bf(Wt[(size_t)(k0 + r) * N + (n0 + c)]);
  }
}

// ---------------------------------------------------------------------------
// GEMM: [M x 512] @ [512 x N] + bias.  Block tile 128x128, 256 threads
// (8 waves: 4 along M x 2 along N; each wave 2x4 WMMA tiles = 32x64).
// Double-buffered K loop: stage slice k+32 while issuing WMMAs for slice k.
// ---------------------------------------------------------------------------
template<int N, bool ABF16, bool OUTBF16>
__global__ __launch_bounds__(256)
void gemm_k512(const float* __restrict__ Af, const unsigned short* __restrict__ Ab,
               const float* __restrict__ Wt, const float* __restrict__ bias,
               float* __restrict__ outF, unsigned short* __restrict__ outB) {
  constexpr int K = 512;
  __shared__ __align__(16) unsigned short sA[2][128 * LDA];   // 2 x 10 KB
  __shared__ __align__(16) unsigned short sB[2][128 * LDB];   // 2 x 10 KB

  const int m0 = blockIdx.y * 128;
  const int n0 = blockIdx.x * 128;
  const int tid = threadIdx.x;
  const int lane = tid & 31;
  const int wv = tid >> 5;
  const int wm = wv & 3;       // 4 waves over M
  const int wn = wv >> 2;      // 2 waves over N

  v8f acc[2][4];
#pragma unroll
  for (int i = 0; i < 2; ++i)
#pragma unroll
    for (int j = 0; j < 4; ++j) acc[i][j] = zero8();

  stage_tiles<N, ABF16>(Af, Ab, Wt, sA[0], sB[0], m0, n0, 0, tid);
  if constexpr (ABF16) wait_async0();
  __syncthreads();

  for (int k0 = 0; k0 < K; k0 += 32) {
    const int cur = (k0 >> 5) & 1;
    if (k0 + 32 < K)
      stage_tiles<N, ABF16>(Af, Ab, Wt, sA[cur ^ 1], sB[cur ^ 1],
                            m0, n0, k0 + 32, tid);

    v16bf a[2], b[4];
#pragma unroll
    for (int mt = 0; mt < 2; ++mt)
      a[mt] = load_fragA(sA[cur] + (wm * 32 + mt * 16) * LDA, LDA);
#pragma unroll
    for (int nt = 0; nt < 4; ++nt)
      b[nt] = load_fragB(sB[cur] + (wn * 64 + nt * 16) * LDB, LDB);
#pragma unroll
    for (int mt = 0; mt < 2; ++mt)
#pragma unroll
      for (int nt = 0; nt < 4; ++nt)
        acc[mt][nt] = wmma_bf16(a[mt], b[nt], acc[mt][nt]);

    if constexpr (ABF16) wait_async0();   // next slice's async copies landed
    __syncthreads();
  }

  // epilogue: C/D layout -> m = r + 8*(lane>>4), n = lane&15
  const int half = lane >> 4, nl = lane & 15;
#pragma unroll
  for (int mt = 0; mt < 2; ++mt) {
#pragma unroll
    for (int nt = 0; nt < 4; ++nt) {
      const int n = n0 + wn * 64 + nt * 16 + nl;
      const float bv = bias[n];
#pragma unroll
      for (int r = 0; r < 8; ++r) {
        const int m = m0 + wm * 32 + mt * 16 + r + half * 8;
        const float v = acc[mt][nt][r] + bv;
        if constexpr (OUTBF16) outB[(size_t)m * N + n] = f2bf(v);
        else                   outF[(size_t)m * N + n] = v;
      }
    }
  }
}

// ---------------------------------------------------------------------------
// Attention: one block per (window, head). 8 waves; wave w owns query rows
// [16w, 16w+16). K staged via async global->LDS DMA; scores live in
// accumulators; softmax via __shfl_xor within 16-lane halves (matches C/D
// layout). P aliased over sK in LDS (52 KB total).
// ---------------------------------------------------------------------------
__global__ __launch_bounds__(256)
void attn_win(const unsigned short* __restrict__ qkv,
              const float* __restrict__ mask,
              const float* __restrict__ bias_table,
              unsigned short* __restrict__ obuf) {
  constexpr int W = 128, NH = 8, HD = 64, C3 = 1536;
  constexpr int LDK = 72;    // sK row pitch (144B -> 16-perm bank map)
  constexpr int LDV = 136;   // sVt row pitch (272B)
  constexpr int LDP = 136;   // sP row pitch

  __shared__ __align__(16) unsigned short sRegion[W * LDP]; // 34 KB; sK aliases front
  __shared__ __align__(16) unsigned short sVt[HD * LDV];    // 17 KB
  __shared__ float sBias[256];                              // 1 KB

  unsigned short* sK = sRegion;   // [128][LDK], used in phase 1
  unsigned short* sP = sRegion;   // [128][LDP], used in phase 2 (after barrier)

  const int h  = blockIdx.x;
  const int wi = blockIdx.y;
  const int tid = threadIdx.x;
  const int lane = tid & 31;
  const int wv = tid >> 5;
  const int half = lane >> 4, nl = lane & 15;

  const size_t base = (size_t)wi * W * C3;

  // stage K rows [tok][d] via async DMA: 1024 x 16B granules, 4 per thread
#pragma unroll
  for (int i = 0; i < 4; ++i) {
    int idx = i * 256 + tid;
    int tok = idx >> 3, d8 = (idx & 7) << 3;
    async_cp16(lds_off_u32(sK + tok * LDK + d8),
               qkv + base + (size_t)tok * C3 + 512 + h * HD + d8);
  }
  // stage V transposed [d][tok] (VALU transpose; DMA cannot transpose)
#pragma unroll
  for (int i = 0; i < 32; ++i) {
    int idx = i * 256 + tid;
    int tok = idx >> 6, d = idx & 63;
    sVt[d * LDV + tok] = qkv[base + (size_t)tok * C3 + 1024 + h * HD + d];
  }
  if (tid < 255) sBias[tid] = bias_table[tid * NH + h];
  wait_async0();
  __syncthreads();

  // S = Q K^T : 16x128 strip per wave, 16 WMMAs (Q frags direct from global)
  v8f acc[8];
#pragma unroll
  for (int t = 0; t < 8; ++t) acc[t] = zero8();
  const unsigned short* qbase = qkv + base + (size_t)(wv * 16) * C3 + h * HD;
#pragma unroll
  for (int ks = 0; ks < HD; ks += 32) {
    v16bf a = load_fragA(qbase + ks, C3);
#pragma unroll
    for (int t = 0; t < 8; ++t) {
      v16bf b = load_fragB(sK + (t * 16) * LDK + ks, LDK);
      acc[t] = wmma_bf16(a, b, acc[t]);
    }
  }

  // scale + relative-position bias + mask, then register softmax
  const float* mp = mask + (size_t)(wi & 63) * W * W;
  float rinv[8];
  int mrow[8];
#pragma unroll
  for (int r = 0; r < 8; ++r) {
    const int m = wv * 16 + r + half * 8;     // global query row
    mrow[r] = m;
    float mx = -3.0e38f;
#pragma unroll
    for (int t = 0; t < 8; ++t) {
      const int n = t * 16 + nl;
      float s = acc[t][r] * 0.125f + sBias[n - m + 127] + mp[m * W + n];
      acc[t][r] = s;
      mx = fmaxf(mx, s);
    }
    for (int d = 1; d < 16; d <<= 1) mx = fmaxf(mx, __shfl_xor(mx, d, 32));
    float sum = 0.0f;
#pragma unroll
    for (int t = 0; t < 8; ++t) {
      float e = __expf(acc[t][r] - mx);
      acc[t][r] = e;
      sum += e;
    }
    for (int d = 1; d < 16; d <<= 1) sum += __shfl_xor(sum, d, 32);
    rinv[r] = 1.0f / sum;
  }

  __syncthreads();   // all waves finished reading sK (aliased by sP)

  // write probabilities to LDS as bf16
#pragma unroll
  for (int r = 0; r < 8; ++r) {
    const int m = mrow[r];
#pragma unroll
    for (int t = 0; t < 8; ++t) {
      const int n = t * 16 + nl;
      sP[m * LDP + n] = f2bf(acc[t][r] * rinv[r]);
    }
  }
  __syncthreads();

  // O = P @ V : 16x64 strip per wave, 16 WMMAs
  v8f oacc[4];
#pragma unroll
  for (int t = 0; t < 4; ++t) oacc[t] = zero8();
#pragma unroll
  for (int ks = 0; ks < W; ks += 32) {
    v16bf a = load_fragA(sP + (wv * 16) * LDP + ks, LDP);
#pragma unroll
    for (int t = 0; t < 4; ++t) {
      v16bf b = load_fragB(sVt + (t * 16) * LDV + ks, LDV);
      oacc[t] = wmma_bf16(a, b, oacc[t]);
    }
  }

  // store O (bf16) head-major into [ (wi*128+q)*512 + h*64 + d ]
  unsigned short* ob = obuf + ((size_t)wi * W) * 512 + h * HD;
#pragma unroll
  for (int t = 0; t < 4; ++t) {
    const int d = t * 16 + nl;
#pragma unroll
    for (int r = 0; r < 8; ++r) {
      const int q = wv * 16 + r + half * 8;
      ob[(size_t)q * 512 + d] = f2bf(oacc[t][r]);
    }
  }
}

// ---------------------------------------------------------------------------
extern "C" void kernel_launch(void* const* d_in, const int* in_sizes, int n_in,
                              void* d_out, int out_size, void* d_ws, size_t ws_size,
                              hipStream_t stream) {
  (void)in_sizes; (void)n_in; (void)out_size; (void)ws_size;
  const float* x          = (const float*)d_in[0];  // (1024,128,512)
  const float* mask       = (const float*)d_in[1];  // (64,128,128)
  const float* w_qkv      = (const float*)d_in[2];  // (512,1536)
  const float* b_qkv      = (const float*)d_in[3];  // (1536)
  const float* w_proj     = (const float*)d_in[4];  // (512,512)
  const float* b_proj     = (const float*)d_in[5];  // (512)
  const float* bias_table = (const float*)d_in[6];  // (255,8)
  float* out = (float*)d_out;                       // (1024,128,512)

  // workspace: qkv bf16 (131072x1536) then attn-out bf16 (131072x512) = 512 MB
  unsigned short* qkv  = (unsigned short*)d_ws;
  unsigned short* obuf = qkv + (size_t)131072 * 1536;

  dim3 blk(256);
  gemm_k512<1536, false, true><<<dim3(12, 1024), blk, 0, stream>>>(
      x, nullptr, w_qkv, b_qkv, nullptr, qkv);
  attn_win<<<dim3(8, 1024), blk, 0, stream>>>(qkv, mask, bias_table, obuf);
  gemm_k512<512, true, false><<<dim3(4, 1024), blk, 0, stream>>>(
      nullptr, obuf, w_proj, b_proj, out, nullptr);
}
